// Model_58695023067699
// MI455X (gfx1250) — matrix operand order repaired and checked by reading, not compiled, not run on hardware
//
#include <hip/hip_runtime.h>
#include <hip/hip_bf16.h>
#include <stdint.h>

#define H 128

typedef __attribute__((ext_vector_type(16))) _Float16 v16h;
typedef __attribute__((ext_vector_type(8)))  float    v8f;

// ---------------------------------------------------------------------------
// Fragment loaders (wave32); A and B use the same contiguous k order
// A (16x32 f16): lane L -> row = L%16 ; half h -> k = kb + 16*(L/16) + h
// B (32x16 f16): lane L -> col = L%16 ; half h -> k = kb + 16*(L/16) + h
// ---------------------------------------------------------------------------
__device__ __forceinline__ v16h load_fragA(const _Float16* __restrict__ rowptr,
                                           int kb, int g) {
    const uint32_t* p = (const uint32_t*)rowptr;   // row of 128 halves
    union { v16h h; uint32_t u[8]; } f;
    int b0 = (kb >> 1) + 8 * g;       // halves  kb+16g   .. kb+16g+7
    int b1 = b0 + 4;                  // halves  kb+16g+8 .. kb+16g+15
#pragma unroll
    for (int i = 0; i < 4; ++i) { f.u[i] = p[b0 + i]; f.u[4 + i] = p[b1 + i]; }
    return f.h;
}

__device__ __forceinline__ v16h load_fragB(const _Float16* __restrict__ colptr,
                                           int kb, int g) {
    const uint32_t* p = (const uint32_t*)colptr;   // transposed-weight row (128 halves)
    union { v16h h; uint32_t u[8]; } f;
    int b0 = (kb >> 1) + 8 * g;       // halves kb+16g .. kb+16g+15
#pragma unroll
    for (int i = 0; i < 8; ++i) f.u[i] = p[b0 + i];
    return f.h;
}

// ---------------------------------------------------------------------------
// Weight transpose + f32->f16 convert: Wt[j*128+k] = W[k*128+j]
// ---------------------------------------------------------------------------
__global__ void wT_f16_kernel(const float* __restrict__ W, _Float16* __restrict__ Wt) {
    int j = blockIdx.x;          // output row (original column)
    int k = threadIdx.x;         // 128 threads
    Wt[j * H + k] = (_Float16)W[k * H + j];
}

// ---------------------------------------------------------------------------
// f32 -> f16 bulk convert
// ---------------------------------------------------------------------------
__global__ void f32_to_f16_kernel(const float* __restrict__ in,
                                  _Float16* __restrict__ out, long long n) {
    long long i = (long long)blockIdx.x * blockDim.x + threadIdx.x;
    if (i < n) out[i] = (_Float16)in[i];
}

// ---------------------------------------------------------------------------
// Edge scatter-add: one thread handles 4 features (float4) of one edge.
// idx = e*32 + q ; q in [0,32) covers 128 features.
// ---------------------------------------------------------------------------
__global__ void scatter_add_kernel(const float* __restrict__ xsrc,
                                   const int* __restrict__ src,
                                   const int* __restrict__ dst,
                                   float* __restrict__ agg,
                                   float* __restrict__ cnt,
                                   int nE, int do_cnt) {
    long long idx = (long long)blockIdx.x * blockDim.x + threadIdx.x;
    int e = (int)(idx >> 5);
    int q = (int)(idx & 31);
    if (e >= nE) return;
    int s = src[e], d = dst[e];
    float4 v = ((const float4*)(xsrc + (size_t)s * H))[q];
    float* ap = agg + (size_t)d * H + q * 4;
    atomicAdd(ap + 0, v.x);
    atomicAdd(ap + 1, v.y);
    atomicAdd(ap + 2, v.z);
    atomicAdd(ap + 3, v.w);
    if (do_cnt && q == 0) atomicAdd(cnt + d, 1.0f);
}

// ---------------------------------------------------------------------------
// mean = agg / max(cnt,1), converted to f16
// ---------------------------------------------------------------------------
__global__ void mean_f16_kernel(const float* __restrict__ agg,
                                const float* __restrict__ cnt,
                                _Float16* __restrict__ out, long long n) {
    long long i = (long long)blockIdx.x * blockDim.x + threadIdx.x;
    if (i >= n) return;
    float c = cnt[i >> 7];
    out[i] = (_Float16)(agg[i] / fmaxf(c, 1.0f));
}

// ---------------------------------------------------------------------------
// SAGE node update via WMMA:
//   out = Aagg @ Wl + Ax @ Wr + bias  (optional ReLU)
// One wave -> one 16x16 tile. blockDim=256 (8 waves = 8 column tiles of H=128),
// gridDim.x = n/16 row tiles (n is a multiple of 16 here).
// ---------------------------------------------------------------------------
__global__ void sage_gemm_wmma(const _Float16* __restrict__ Aagg,   // [n,128] f16
                               const _Float16* __restrict__ Ax,     // [n,128] f16
                               const _Float16* __restrict__ WlT,    // [128,128] f16 (transposed)
                               const _Float16* __restrict__ WrT,    // [128,128] f16 (transposed)
                               const float* __restrict__ bias,      // [128]
                               float* __restrict__ outF,            // [n,128] f32
                               _Float16* __restrict__ outH,         // [n,128] f16 or null
                               int n, int relu) {
    int wave  = threadIdx.x >> 5;            // 0..7 -> column tile
    int lane  = threadIdx.x & 31;
    int g     = lane >> 4;                   // half-wave group
    int r     = lane & 15;
    int mbase = blockIdx.x * 16;
    int nbase = wave * 16;

    int row = mbase + r;
    if (row >= n) row = n - 1;               // safety; n % 16 == 0 in this model

    const _Float16* aggRow = Aagg + (size_t)row * H;
    const _Float16* xRow   = Ax   + (size_t)row * H;
    const _Float16* wlCol  = WlT  + (size_t)(nbase + r) * H;
    const _Float16* wrCol  = WrT  + (size_t)(nbase + r) * H;

    v8f c = {};
#pragma unroll
    for (int kb = 0; kb < H; kb += 32) {
        v16h a0 = load_fragA(aggRow, kb, g);
        v16h b0 = load_fragB(wlCol,  kb, g);
        c = __builtin_amdgcn_wmma_f32_16x16x32_f16(false, a0, false, b0,
                                                   (short)0, c, false, false);
        v16h a1 = load_fragA(xRow,  kb, g);
        v16h b1 = load_fragB(wrCol, kb, g);
        c = __builtin_amdgcn_wmma_f32_16x16x32_f16(false, a1, false, b1,
                                                   (short)0, c, false, false);
    }

    // C/D layout: element v of lane L -> row = mbase + v + 8*(L/16), col = nbase + L%16
    float bv = bias[nbase + r];
#pragma unroll
    for (int v = 0; v < 8; ++v) {
        int orow = mbase + v + 8 * g;
        if (orow >= n) continue;
        float val = c[v] + bv;
        if (relu) val = fmaxf(val, 0.0f);
        outF[(size_t)orow * H + nbase + r] = val;
        if (outH) outH[(size_t)orow * H + nbase + r] = (_Float16)val;
    }
}

// ---------------------------------------------------------------------------
// Edge dot product: one wave per label edge; float4 per lane + shuffle reduce.
// ---------------------------------------------------------------------------
__global__ void edge_dot_kernel(const float* __restrict__ ze,
                                const float* __restrict__ zt,
                                const int* __restrict__ ls,
                                const int* __restrict__ ld,
                                float* __restrict__ out, int nL) {
    int wid  = (int)(((long long)blockIdx.x * blockDim.x + threadIdx.x) >> 5);
    int lane = threadIdx.x & 31;
    if (wid >= nL) return;
    const float4 a = ((const float4*)(ze + (size_t)ls[wid] * H))[lane];
    const float4 b = ((const float4*)(zt + (size_t)ld[wid] * H))[lane];
    float s = a.x * b.x + a.y * b.y + a.z * b.z + a.w * b.w;
#pragma unroll
    for (int off = 16; off > 0; off >>= 1) s += __shfl_down(s, off, 32);
    if (lane == 0) out[wid] = s;
}

// ---------------------------------------------------------------------------
// Host orchestration
// ---------------------------------------------------------------------------
extern "C" void kernel_launch(void* const* d_in, const int* in_sizes, int n_in,
                              void* d_out, int out_size, void* d_ws, size_t ws_size,
                              hipStream_t stream) {
    const float* x_expert = (const float*)d_in[0];
    const float* x_team   = (const float*)d_in[1];
    const int* src_expert = (const int*)d_in[2];
    const int* dst_team   = (const int*)d_in[3];
    const int* label_src  = (const int*)d_in[4];
    const int* label_dst  = (const int*)d_in[5];
    const float* Wf[8];
    for (int i = 0; i < 8; ++i) Wf[i] = (const float*)d_in[6 + i];
    const float* b1et = (const float*)d_in[14];
    const float* b1te = (const float*)d_in[15];
    const float* b2et = (const float*)d_in[16];
    const float* b2te = (const float*)d_in[17];

    const int nE = in_sizes[0] / H;      // experts
    const int nT = in_sizes[1] / H;      // teams
    const int nEdge  = in_sizes[2];
    const int nLabel = in_sizes[4];

    // ---- workspace layout ----
    char* w = (char*)d_ws;
    size_t off = 0;
    auto alloc = [&](size_t bytes) -> void* {
        void* p = w + off;
        off = (off + bytes + 255) & ~(size_t)255;
        return p;
    };
    _Float16* Wt[8];
    for (int i = 0; i < 8; ++i) Wt[i] = (_Float16*)alloc((size_t)H * H * sizeof(_Float16));
    float* agg_t  = (float*)alloc((size_t)nT * H * sizeof(float));
    float* agg_e  = (float*)alloc((size_t)nE * H * sizeof(float));
    float* cnt_t  = (float*)alloc((size_t)nT * sizeof(float));
    float* cnt_e  = (float*)alloc((size_t)nE * sizeof(float));
    _Float16* mean_t = (_Float16*)alloc((size_t)nT * H * sizeof(_Float16));
    _Float16* mean_e = (_Float16*)alloc((size_t)nE * H * sizeof(_Float16));
    _Float16* xT_h   = (_Float16*)alloc((size_t)nT * H * sizeof(_Float16));
    _Float16* xE_h   = (_Float16*)alloc((size_t)nE * H * sizeof(_Float16));
    float* h_team    = (float*)alloc((size_t)nT * H * sizeof(float));
    float* h_expert  = (float*)alloc((size_t)nE * H * sizeof(float));
    _Float16* hT_h   = (_Float16*)alloc((size_t)nT * H * sizeof(_Float16));
    _Float16* hE_h   = (_Float16*)alloc((size_t)nE * H * sizeof(_Float16));
    float* z_team    = (float*)alloc((size_t)nT * H * sizeof(float));
    float* z_expert  = (float*)alloc((size_t)nE * H * sizeof(float));
    (void)ws_size; (void)n_in; (void)out_size;

    auto cdiv = [](long long a, long long b) { return (int)((a + b - 1) / b); };

    // ---- weight prep ----
    for (int i = 0; i < 8; ++i)
        wT_f16_kernel<<<H, H, 0, stream>>>(Wf[i], Wt[i]);

    // ---- feature converts ----
    f32_to_f16_kernel<<<cdiv((long long)nT * H, 256), 256, 0, stream>>>(x_team, xT_h, (long long)nT * H);
    f32_to_f16_kernel<<<cdiv((long long)nE * H, 256), 256, 0, stream>>>(x_expert, xE_h, (long long)nE * H);

    // ---- layer 1 aggregation ----
    hipMemsetAsync(agg_t, 0, (size_t)nT * H * sizeof(float), stream);
    hipMemsetAsync(agg_e, 0, (size_t)nE * H * sizeof(float), stream);
    hipMemsetAsync(cnt_t, 0, (size_t)nT * sizeof(float), stream);
    hipMemsetAsync(cnt_e, 0, (size_t)nE * sizeof(float), stream);

    int sblocks = cdiv((long long)nEdge * 32, 256);
    scatter_add_kernel<<<sblocks, 256, 0, stream>>>(x_expert, src_expert, dst_team, agg_t, cnt_t, nEdge, 1);
    scatter_add_kernel<<<sblocks, 256, 0, stream>>>(x_team, dst_team, src_expert, agg_e, cnt_e, nEdge, 1);

    mean_f16_kernel<<<cdiv((long long)nT * H, 256), 256, 0, stream>>>(agg_t, cnt_t, mean_t, (long long)nT * H);
    mean_f16_kernel<<<cdiv((long long)nE * H, 256), 256, 0, stream>>>(agg_e, cnt_e, mean_e, (long long)nE * H);

    // ---- layer 1 GEMMs (ReLU) ----
    sage_gemm_wmma<<<cdiv(nT, 16), 256, 0, stream>>>(mean_t, xT_h, Wt[0], Wt[1], b1et,
                                                     h_team, hT_h, nT, 1);   // W1et_l/r
    sage_gemm_wmma<<<cdiv(nE, 16), 256, 0, stream>>>(mean_e, xE_h, Wt[2], Wt[3], b1te,
                                                     h_expert, hE_h, nE, 1); // W1te_l/r

    // ---- layer 2 aggregation (counts reused) ----
    hipMemsetAsync(agg_t, 0, (size_t)nT * H * sizeof(float), stream);
    hipMemsetAsync(agg_e, 0, (size_t)nE * H * sizeof(float), stream);
    scatter_add_kernel<<<sblocks, 256, 0, stream>>>(h_expert, src_expert, dst_team, agg_t, cnt_t, nEdge, 0);
    scatter_add_kernel<<<sblocks, 256, 0, stream>>>(h_team, dst_team, src_expert, agg_e, cnt_e, nEdge, 0);

    mean_f16_kernel<<<cdiv((long long)nT * H, 256), 256, 0, stream>>>(agg_t, cnt_t, mean_t, (long long)nT * H);
    mean_f16_kernel<<<cdiv((long long)nE * H, 256), 256, 0, stream>>>(agg_e, cnt_e, mean_e, (long long)nE * H);

    // ---- layer 2 GEMMs (no activation) ----
    sage_gemm_wmma<<<cdiv(nT, 16), 256, 0, stream>>>(mean_t, hT_h, Wt[4], Wt[5], b2et,
                                                     z_team, (_Float16*)nullptr, nT, 0);   // W2et_l/r
    sage_gemm_wmma<<<cdiv(nE, 16), 256, 0, stream>>>(mean_e, hE_h, Wt[6], Wt[7], b2te,
                                                     z_expert, (_Float16*)nullptr, nE, 0); // W2te_l/r

    // ---- edge dot classifier ----
    edge_dot_kernel<<<cdiv((long long)nLabel * 32, 256), 256, 0, stream>>>(
        z_expert, z_team, label_src, label_dst, (float*)d_out, nLabel);
}
